// FastRCNN_84997402788324
// MI455X (gfx1250) — matrix operand order repaired and checked by reading, not compiled, not run on hardware
//
#include <hip/hip_runtime.h>
#include <hip/hip_bf16.h>

typedef _Float16 v16h __attribute__((ext_vector_type(16)));
typedef _Float16 v8h  __attribute__((ext_vector_type(8)));
typedef float    v8f  __attribute__((ext_vector_type(8)));
typedef unsigned int u32x4 __attribute__((ext_vector_type(4)));
typedef int          i32x4 __attribute__((ext_vector_type(4)));
typedef int          i32x8 __attribute__((ext_vector_type(8)));

#define PP 7
#define NCLS 21
#define N_ROI 1000
#define CCH 512
#define KDIM (CCH * PP * PP)   /* 25088 */
#define HDIM 1024
#define NCAND (N_ROI * (NCLS - 1)) /* 20000 */
#define NDETS 100

#if __has_builtin(__builtin_amdgcn_tensor_load_to_lds)
#define USE_TDM 1
#else
#define USE_TDM 0
#endif

// ---------------------------------------------------------------- transpose+cvt
__global__ void cvt_transpose_f16(const float* __restrict__ src, _Float16* __restrict__ dst,
                                  int K, int N) {
    long idx = (long)blockIdx.x * blockDim.x + threadIdx.x;
    long total = (long)K * N;
    if (idx >= total) return;
    int k = (int)(idx / N);
    int n = (int)(idx % N);
    dst[(long)n * K + k] = (_Float16)src[idx];
}

// ---------------------------------------------------------------- ROI max pool
__global__ void roi_pool_kernel(const float* __restrict__ fmap, const float* __restrict__ rois,
                                _Float16* __restrict__ X) {
    int roi = blockIdx.x;
    const float* r = rois + roi * 4;
    float x1 = roundf(r[0] * 0.03125f);
    float y1 = roundf(r[1] * 0.03125f);
    float x2 = roundf(r[2] * 0.03125f);
    float y2 = roundf(r[3] * 0.03125f);
    float rw = fmaxf(x2 - x1 + 1.0f, 1.0f);
    float rh = fmaxf(y2 - y1 + 1.0f, 1.0f);
    float bw = rw / 7.0f, bh = rh / 7.0f;
    for (int col = threadIdx.x; col < KDIM; col += blockDim.x) {
        int c = col / 49;
        int bin = col % 49;
        int ph = bin / 7, pw = bin % 7;
        float hs = fminf(fmaxf(floorf((float)ph * bh) + y1, 0.0f), 16.0f);
        float he = fminf(fmaxf(ceilf((float)(ph + 1) * bh) + y1, 0.0f), 16.0f);
        float wl = fminf(fmaxf(floorf((float)pw * bw) + x1, 0.0f), 16.0f);
        float wr = fminf(fmaxf(ceilf((float)(pw + 1) * bw) + x1, 0.0f), 16.0f);
        int ys0 = (int)hs, ys1 = (int)he, xs0 = (int)wl, xs1 = (int)wr;
        float v = 0.0f;
        if (ys0 < ys1 && xs0 < xs1) {
            float m = -1e30f;
            const float* fc = fmap + (long)c * 256;
            for (int y = ys0; y < ys1; ++y)
                for (int x = xs0; x < xs1; ++x)
                    m = fmaxf(m, fc[y * 16 + x]);
            v = m;
        }
        X[(long)roi * KDIM + col] = (_Float16)v;
    }
}

// ---------------------------------------------------------------- TDM helper
// Loads a 128-row x 32-halfword tile (row stride K halfwords) from gptr into
// LDS at lds_addr with padded row stride 40 halfwords (16 DW data + 4 DW pad).
__device__ __forceinline__ void tdm_load_tile_b(unsigned lds_addr, const _Float16* gptr, int K) {
#if USE_TDM
    unsigned long long ga = (unsigned long long)(uintptr_t)gptr;
    u32x4 g0 = { 1u,                                   // count=1, user descriptor
                 lds_addr,                             // LDS byte address
                 (unsigned)(ga & 0xFFFFFFFFu),
                 (unsigned)((ga >> 32) & 0x01FFFFFFu) | 0x80000000u };  // type=2
    i32x8 g1 = { (int)((1u << 16)                      // data_size = 2 bytes
                     | (1u << 20)                      // pad_enable
                     | (3u << 22)                      // pad_interval: 16 DWORDs
                     | (3u << 25)),                    // pad_amount: 4 DWORDs
                 (int)(32u << 16),                     // tensor_dim0 = 32 (lo16)
                 (int)(128u << 16),                    // tensor_dim1 = 128 (lo16)
                 (int)(32u << 16),                     // tile_dim0 = 32
                 128,                                  // tile_dim1 = 128, tile_dim2 = 0
                 K,                                    // tensor_dim0_stride lo32
                 0, 0 };
    i32x4 z4 = { 0, 0, 0, 0 };
#if __clang_major__ >= 23
    i32x8 z8 = { 0, 0, 0, 0, 0, 0, 0, 0 };
    __builtin_amdgcn_tensor_load_to_lds(g0, g1, z4, z4, z8, 0);
#else
    __builtin_amdgcn_tensor_load_to_lds(g0, g1, z4, z4, 0);
#endif
#endif
}

// ---------------------------------------------------------------- WMMA GEMM
// C[M][N] = relu( A[M][K] * B[N][K]^T + bias[N] ), f16 in / f32 acc / f16 out.
__global__ __launch_bounds__(256)
void gemm_f16_wmma(const _Float16* __restrict__ A, const _Float16* __restrict__ B,
                   const float* __restrict__ bias, _Float16* __restrict__ C,
                   int M, int N, int K, int relu) {
    __shared__ _Float16 As[128][40];   // 128 rows x 32 k (+pad)
    __shared__ _Float16 Bs[128][40];   // 128 cols x 32 k (+pad), TDM-padded layout
    int tid  = threadIdx.x;
    int lane = tid & 31;
    int wave = tid >> 5;          // 0..7
    int wm = wave & 3;            // 4 m-groups of 32 rows
    int wn = wave >> 2;           // 2 n-groups of 64 cols
    long bm = (long)blockIdx.y * 128;
    long bn = (long)blockIdx.x * 128;
    int ldr = tid >> 1;           // tile row 0..127
    int ldc = (tid & 1) * 16;     // k-offset 0 or 16

    v8f acc[2][4] = {};
    int r  = lane & 15;
    int hi = lane >> 4;
    int aoff = hi * 8;            // A-frag base k: 0 / 8
    int boff = hi * 8;            // B-frag base k: 0 / 8

    // clamp instead of branch: padded rows only feed masked-out outputs
    long arow = bm + ldr;
    if (arow >= M) arow = M - 1;
    const _Float16* pa_base = A + arow * K + ldc;
    unsigned bs_lds = (unsigned)(uintptr_t)&Bs[0][0];
    const _Float16* pb_tile0 = B + bn * K;

    for (int kt = 0; kt < K; kt += 32) {
        const _Float16* pa = pa_base + kt;
        v8h a0 = *(const v8h*)pa;
        v8h a1 = *(const v8h*)(pa + 8);
#if !USE_TDM
        long brow = bn + ldr;
        const _Float16* pb = B + brow * K + kt + ldc;
        v8h b0 = *(const v8h*)pb, b1 = *(const v8h*)(pb + 8);
#endif
        if (kt + 32 < K) __builtin_prefetch(pa + 32, 0, 1);  // global_prefetch_b8
        __syncthreads();
        if (wave == 0) tdm_load_tile_b(bs_lds, pb_tile0 + kt, K);  // async tensor DMA
        *(v8h*)&As[ldr][ldc] = a0;  *(v8h*)&As[ldr][ldc + 8] = a1;
#if !USE_TDM
        *(v8h*)&Bs[ldr][ldc] = b0;  *(v8h*)&Bs[ldr][ldc + 8] = b1;
#endif
        if (wave == 0) __builtin_amdgcn_s_wait_tensorcnt(0);
        __syncthreads();

        v16h af[2], bf[4];
        for (int i = 0; i < 2; ++i) {            // ISA 16-bit A 16x32 layout
            int m = wm * 32 + i * 16 + r;
            v8h lo = *(const v8h*)&As[m][aoff];
            v8h hh = *(const v8h*)&As[m][aoff + 16];
            for (int e = 0; e < 8; ++e) { af[i][e] = lo[e]; af[i][e + 8] = hh[e]; }
        }
        for (int j = 0; j < 4; ++j) {            // ISA 16-bit B 32x16 layout
            int n = wn * 64 + j * 16 + r;
            v8h lo = *(const v8h*)&Bs[n][boff];
            v8h hh = *(const v8h*)&Bs[n][boff + 16];
            for (int e = 0; e < 8; ++e) { bf[j][e] = lo[e]; bf[j][e + 8] = hh[e]; }
        }
        for (int i = 0; i < 2; ++i)
            for (int j = 0; j < 4; ++j)
                acc[i][j] = __builtin_amdgcn_wmma_f32_16x16x32_f16(
                    false, af[i], false, bf[j], (short)0, acc[i][j], false, false);
    }

    for (int i = 0; i < 2; ++i)
        for (int j = 0; j < 4; ++j) {
            long col = bn + wn * 64 + j * 16 + r;
            float bv = bias[col];
            for (int e = 0; e < 8; ++e) {
                long row = bm + wm * 32 + i * 16 + hi * 8 + e;
                if (row < M) {
                    float v = acc[i][j][e] + bv;
                    if (relu) v = fmaxf(v, 0.0f);
                    C[row * N + col] = (_Float16)v;
                }
            }
        }
}

// ---------------------------------------------------------------- per-ROI head
__global__ __launch_bounds__(128)
void head_kernel(const _Float16* __restrict__ H2, const _Float16* __restrict__ Wct,
                 const _Float16* __restrict__ Wbt, const float* __restrict__ bc,
                 const float* __restrict__ bb, const float* __restrict__ rois,
                 float* __restrict__ cboxes, float* __restrict__ coff,
                 float* __restrict__ wsS) {
    __shared__ float h[HDIM];
    __shared__ float probs[NCLS];
    __shared__ float regs[NCLS * 4];
    int roi = blockIdx.x, tid = threadIdx.x;
    for (int k = tid; k < HDIM; k += 128) h[k] = (float)H2[(long)roi * HDIM + k];
    __syncthreads();
    if (tid < 105) {
        const _Float16* w = (tid < 21) ? (Wct + (long)tid * HDIM)
                                       : (Wbt + (long)(tid - 21) * HDIM);
        float s = 0.0f;
        for (int k = 0; k < HDIM; ++k) s += h[k] * (float)w[k];
        if (tid < 21) probs[tid] = s + bc[tid];
        else          regs[tid - 21] = s + bb[tid - 21];
    }
    __syncthreads();
    if (tid == 0) {
        float mx = probs[0];
        for (int c = 1; c < NCLS; ++c) mx = fmaxf(mx, probs[c]);
        float sum = 0.0f;
        for (int c = 0; c < NCLS; ++c) { float e = expf(probs[c] - mx); probs[c] = e; sum += e; }
        float inv = 1.0f / sum;
        for (int c = 0; c < NCLS; ++c) probs[c] *= inv;
    }
    __syncthreads();
    if (tid >= 1 && tid < NCLS) {
        int c = tid;
        const float* r = rois + roi * 4;
        float x1 = r[0], y1 = r[1], x2 = r[2], y2 = r[3];
        float w  = x2 - x1, hh = y2 - y1;
        float cx = x1 + 0.5f * w, cy = y1 + 0.5f * hh;
        float dx = regs[c * 4 + 0], dy = regs[c * 4 + 1];
        float dw = regs[c * 4 + 2], dh = regs[c * 4 + 3];
        float pcx = dx * w + cx, pcy = dy * hh + cy;
        float pw = expf(dw) * w, phh = expf(dh) * hh;
        float bx1 = fminf(fmaxf(pcx - 0.5f * pw,  0.0f), 512.0f);
        float by1 = fminf(fmaxf(pcy - 0.5f * phh, 0.0f), 512.0f);
        float bx2 = fminf(fmaxf(pcx + 0.5f * pw,  0.0f), 512.0f);
        float by2 = fminf(fmaxf(pcy + 0.5f * phh, 0.0f), 512.0f);
        int idx = roi * 20 + (c - 1);
        cboxes[idx * 4 + 0] = bx1; cboxes[idx * 4 + 1] = by1;
        cboxes[idx * 4 + 2] = bx2; cboxes[idx * 4 + 3] = by2;
        float off = (float)c * 10000.0f;
        coff[idx * 4 + 0] = bx1 + off; coff[idx * 4 + 1] = by1 + off;
        coff[idx * 4 + 2] = bx2 + off; coff[idx * 4 + 3] = by2 + off;
        float sc = probs[c];
        bool valid = (sc > 0.1f) && ((bx2 - bx1) >= 0.01f) && ((by2 - by1) >= 0.01f);
        wsS[idx] = valid ? sc : -1.0f;
    }
}

// ---------------------------------------------------------------- NMS
__global__ __launch_bounds__(256)
void nms_kernel(const float* __restrict__ cboxes, const float* __restrict__ coff,
                float* __restrict__ wsS, float* __restrict__ out) {
    __shared__ float sv[256];
    __shared__ int   si[256];
    __shared__ float bjs[4];
    int tid = threadIdx.x;
    for (int d = 0; d < NDETS; ++d) {
        float best = -2.0f; int bi = 0;
        for (int i = tid; i < NCAND; i += 256) {
            float s = wsS[i];
            if (s > best) { best = s; bi = i; }
        }
        sv[tid] = best; si[tid] = bi;
        __syncthreads();
        for (int o = 128; o > 0; o >>= 1) {
            if (tid < o) {
                float so = sv[tid + o]; int io = si[tid + o];
                if (so > sv[tid] || (so == sv[tid] && io < si[tid])) { sv[tid] = so; si[tid] = io; }
            }
            __syncthreads();
        }
        int j = si[0]; float sj = sv[0];
        bool ok = sj > 0.0f;
        if (tid == 0) {
            for (int t = 0; t < 4; ++t) out[d * 4 + t] = ok ? cboxes[j * 4 + t] : 0.0f;
            out[400 + d] = ok ? sj : 0.0f;
            ((int*)out)[500 + d] = ok ? (j % 20 + 1) : 0;
            bjs[0] = coff[j * 4 + 0]; bjs[1] = coff[j * 4 + 1];
            bjs[2] = coff[j * 4 + 2]; bjs[3] = coff[j * 4 + 3];
        }
        __syncthreads();
        float bx1 = bjs[0], by1 = bjs[1], bx2 = bjs[2], by2 = bjs[3];
        float aj = (bx2 - bx1) * (by2 - by1);
        for (int i = tid; i < NCAND; i += 256) {
            float ix1 = coff[i * 4 + 0], iy1 = coff[i * 4 + 1];
            float ix2 = coff[i * 4 + 2], iy2 = coff[i * 4 + 3];
            float w = fmaxf(fminf(ix2, bx2) - fmaxf(ix1, bx1), 0.0f);
            float h = fmaxf(fminf(iy2, by2) - fmaxf(iy1, by1), 0.0f);
            float inter = w * h;
            float ai = (ix2 - ix1) * (iy2 - iy1);
            float iou = inter / fmaxf(ai + aj - inter, 1e-6f);
            if (iou > 0.3f) wsS[i] = -1.0f;
        }
        __syncthreads();
    }
}

// ---------------------------------------------------------------- launch
extern "C" void kernel_launch(void* const* d_in, const int* in_sizes, int n_in,
                              void* d_out, int out_size, void* d_ws, size_t ws_size,
                              hipStream_t stream) {
    const float* fmap = (const float*)d_in[0];
    const float* rois = (const float*)d_in[1];
    const float* W1   = (const float*)d_in[2];
    const float* b1   = (const float*)d_in[3];
    const float* W2   = (const float*)d_in[4];
    const float* b2   = (const float*)d_in[5];
    const float* Wc   = (const float*)d_in[6];
    const float* bc   = (const float*)d_in[7];
    const float* Wb   = (const float*)d_in[8];
    const float* bb   = (const float*)d_in[9];

    char* ws = (char*)d_ws;
    size_t off = 0;
    auto alloc = [&](size_t bytes) {
        size_t o = off;
        off = (off + bytes + 255) & ~(size_t)255;
        return o;
    };
    _Float16* X    = (_Float16*)(ws + alloc((size_t)N_ROI * KDIM * 2));
    _Float16* W1t  = (_Float16*)(ws + alloc((size_t)KDIM * HDIM * 2));
    _Float16* H1   = (_Float16*)(ws + alloc((size_t)N_ROI * HDIM * 2));
    _Float16* W2t  = (_Float16*)(ws + alloc((size_t)HDIM * HDIM * 2));
    _Float16* H2   = (_Float16*)(ws + alloc((size_t)N_ROI * HDIM * 2));
    _Float16* Wct  = (_Float16*)(ws + alloc((size_t)NCLS * HDIM * 2));
    _Float16* Wbt  = (_Float16*)(ws + alloc((size_t)NCLS * 4 * HDIM * 2));
    float*    cbx  = (float*)(ws + alloc((size_t)NCAND * 4 * 4));
    float*    coff = (float*)(ws + alloc((size_t)NCAND * 4 * 4));
    float*    wsS  = (float*)(ws + alloc((size_t)NCAND * 4));

    // weights -> f16, transposed to [N][K]
    {
        long t1 = (long)KDIM * HDIM;
        cvt_transpose_f16<<<dim3((unsigned)((t1 + 255) / 256)), 256, 0, stream>>>(W1, W1t, KDIM, HDIM);
        long t2 = (long)HDIM * HDIM;
        cvt_transpose_f16<<<dim3((unsigned)((t2 + 255) / 256)), 256, 0, stream>>>(W2, W2t, HDIM, HDIM);
        long t3 = (long)HDIM * NCLS;
        cvt_transpose_f16<<<dim3((unsigned)((t3 + 255) / 256)), 256, 0, stream>>>(Wc, Wct, HDIM, NCLS);
        long t4 = (long)HDIM * NCLS * 4;
        cvt_transpose_f16<<<dim3((unsigned)((t4 + 255) / 256)), 256, 0, stream>>>(Wb, Wbt, HDIM, NCLS * 4);
    }

    roi_pool_kernel<<<N_ROI, 256, 0, stream>>>(fmap, rois, X);

    // FC1: [1000 x 25088] x [25088 x 1024] -> relu -> H1
    gemm_f16_wmma<<<dim3(HDIM / 128, (N_ROI + 127) / 128), 256, 0, stream>>>(
        X, W1t, b1, H1, N_ROI, HDIM, KDIM, 1);
    // FC2: [1000 x 1024] x [1024 x 1024] -> relu -> H2
    gemm_f16_wmma<<<dim3(HDIM / 128, (N_ROI + 127) / 128), 256, 0, stream>>>(
        H1, W2t, b2, H2, N_ROI, HDIM, HDIM, 1);

    head_kernel<<<N_ROI, 128, 0, stream>>>(H2, Wct, Wbt, bc, bb, rois, cbx, coff, wsS);
    nms_kernel<<<1, 256, 0, stream>>>(cbx, coff, wsS, (float*)d_out);
}
